// SelfAttention_11012296147292
// MI455X (gfx1250) — matrix operand order repaired and checked by reading, not compiled, not run on hardware
//
#include <hip/hip_runtime.h>
#include <hip/hip_bf16.h>

typedef _Float16 f16;
typedef __attribute__((ext_vector_type(16))) _Float16 v16h;
typedef __attribute__((ext_vector_type(8)))  _Float16 v8h;
typedef __attribute__((ext_vector_type(8)))  float    v8f;
typedef __attribute__((ext_vector_type(4)))  float    v4f;
typedef __attribute__((ext_vector_type(4)))  int      v4i;

#define SEQ    2048
#define DIM    128
#define QTILE  128     // q rows per workgroup
#define KTILE  32      // keys per inner block (== WMMA K for P@V)
#define NWAVES 8       // 8 wave32 = 256 threads
#define NKBLK  (SEQ / KTILE)

// ---------------- CDNA5 async-to-LDS path (guarded) ----------------
#if defined(__has_builtin)
#if __has_builtin(__builtin_amdgcn_global_load_async_to_lds_b128) && \
    __has_builtin(__builtin_amdgcn_s_wait_asynccnt)
#define FA_ASYNC_LDS 1
#endif
#endif

// Pointer-to-addrspace-qualified-v4i types (attribute binds to the pointee).
typedef __attribute__((address_space(1))) v4i* gp_v4i;
typedef __attribute__((address_space(3))) v4i* lp_v4i;

__device__ __forceinline__ void cp16_g2lds(const f16* gsrc, f16* ldst) {
#ifdef FA_ASYNC_LDS
  __builtin_amdgcn_global_load_async_to_lds_b128((gp_v4i)gsrc, (lp_v4i)ldst,
                                                 /*offset=*/0, /*cpol=*/0);
#else
  *(v8h*)ldst = *(const v8h*)gsrc;   // synchronous fallback, still f16 bytes
#endif
}
__device__ __forceinline__ void wait_async_all() {
#ifdef FA_ASYNC_LDS
  __builtin_amdgcn_s_wait_asynccnt(0);
#endif
}

// Butterfly reductions within a 16-lane half of the wave (C-layout rows live
// across lanes 0-15 / 16-31; xor masks < 16 never cross the boundary).
__device__ __forceinline__ float rowmax16(float v) {
  v = fmaxf(v, __shfl_xor(v, 1, 32));
  v = fmaxf(v, __shfl_xor(v, 2, 32));
  v = fmaxf(v, __shfl_xor(v, 4, 32));
  v = fmaxf(v, __shfl_xor(v, 8, 32));
  return v;
}
__device__ __forceinline__ float rowsum16(float v) {
  v += __shfl_xor(v, 1, 32);
  v += __shfl_xor(v, 2, 32);
  v += __shfl_xor(v, 4, 32);
  v += __shfl_xor(v, 8, 32);
  return v;
}

// Load a 16-half fragment from LDS as two 16-byte pieces (gap in halves
// between the two pieces: 8 => fully contiguous, used for every fragment here, P included).
__device__ __forceinline__ v16h ld_frag(const f16* p, int gap) {
  v8h lo = *(const v8h*)(p);
  v8h hi = *(const v8h*)(p + gap);
  v16h r;
#pragma unroll
  for (int j = 0; j < 8; ++j) { r[j] = lo[j]; r[j + 8] = hi[j]; }
  return r;
}

// ---------------------------------------------------------------------------
// One-time preprocessing: K -> f16 row-major, V -> f16 block-tiled transpose
//   VTh[((b*NKBLK + kb)*DIM + d)*KTILE + kk] = V[b][kb*32+kk][d]
// so each key block's V^T tile is one contiguous 8KB slab (async-copy ready).
// ---------------------------------------------------------------------------
__global__ void cvt_kv_kernel(const float* __restrict__ K,
                              const float* __restrict__ V,
                              f16* __restrict__ Kh,
                              f16* __restrict__ VTh, int total) {
  for (int i = blockIdx.x * blockDim.x + threadIdx.x; i < total;
       i += gridDim.x * blockDim.x) {
    Kh[i] = (f16)K[i];
    const int d  = i & (DIM - 1);
    const int s  = (i / DIM) & (SEQ - 1);
    const int bb = i / (DIM * SEQ);
    const int kb = s >> 5, kk = s & 31;
    VTh[(((size_t)bb * NKBLK + kb) * DIM + d) * KTILE + kk] = (f16)V[i];
  }
}

// ---------------------------------------------------------------------------
// Flash attention, f16 WMMA, double-buffered async LDS staging.
// ---------------------------------------------------------------------------
__global__ __launch_bounds__(256)
void fa_fwd_async(const float* __restrict__ Q,
                  const f16* __restrict__ Kh,
                  const f16* __restrict__ VTh,
                  float* __restrict__ O) {
  __shared__ f16 lds_k [2][KTILE * DIM];       // K block, [key][d] f16
  __shared__ f16 lds_vt[2][DIM * KTILE];       // V^T block, [d][key] f16
  __shared__ f16 lds_p [NWAVES * 16 * KTILE];  // per-wave P tile [16][32] f16

  const int qblk  = blockIdx.x;
  const int b     = blockIdx.y;
  const int qbase = qblk * QTILE;
  const int tid   = threadIdx.x;
  const int wave  = tid >> 5;
  const int lane  = tid & 31;
  const int lhalf = lane >> 4;
  const int l16   = lane & 15;
  const int qrow0 = qbase + wave * 16;

  const float* Qb   = Q   + (size_t)b * SEQ * DIM;
  const f16*   Kb16 = Kh  + (size_t)b * SEQ * DIM;           // row-major
  const f16*   VTb  = VTh + (size_t)b * NKBLK * DIM * KTILE; // block-tiled
  float*       Ob   = O   + (size_t)b * SEQ * DIM;

  // Q A-fragments, contiguous k order, same as the B fragments:
  //   lane: M = lane&15;  K(e) = c*32 + (lane>=16?16:0) + e
  v16h qfrag[4];
  {
    const float* qr = Qb + (size_t)(qrow0 + l16) * DIM;
#pragma unroll
    for (int c = 0; c < 4; ++c) {
      const int d0 = c * 32 + lhalf * 16;
      v16h a;
#pragma unroll
      for (int e = 0; e < 8; ++e) {
        a[e]     = (f16)qr[d0 + e];
        a[e + 8] = (f16)qr[d0 + 8 + e];
      }
      qfrag[c] = a;
    }
  }

  float mrow[8], lrow[8];
  v8f   oacc[8];
#pragma unroll
  for (int r = 0; r < 8; ++r) { mrow[r] = -__builtin_inff(); lrow[r] = 0.f; }
#pragma unroll
  for (int t = 0; t < 8; ++t) { oacc[t] = {}; }

  // exp2-domain softmax: scores scaled by (1/sqrt(128))*log2(e)
  const float scale2 = 0.08838834764831845f * 1.4426950408889634f;
  const int   nkb    = (qbase + QTILE) / KTILE;  // causal

  f16* pw = lds_p + wave * 16 * KTILE;

  // prologue: stage block 0 into buffer 0 (2 x 16B per thread per matrix)
#pragma unroll
  for (int i = 0; i < 2; ++i) {
    const int h = (tid + i * 256) * 8;
    cp16_g2lds(Kb16 + h, &lds_k[0][h]);
    cp16_g2lds(VTb  + h, &lds_vt[0][h]);
  }

  for (int kb = 0; kb < nkb; ++kb) {
    const int key0 = kb * KTILE;
    const int buf  = kb & 1;

    wait_async_all();     // my async writes for block kb have landed
    __syncthreads();      // everyone's writes landed; prev reads of buf^1 done

    if (kb + 1 < nkb) {   // issue next block into the other buffer
      const f16* kg = Kb16 + (size_t)(kb + 1) * KTILE * DIM;
      const f16* vg = VTb  + (size_t)(kb + 1) * DIM * KTILE;
#pragma unroll
      for (int i = 0; i < 2; ++i) {
        const int h = (tid + i * 256) * 8;
        cp16_g2lds(kg + h, &lds_k[buf ^ 1][h]);
        cp16_g2lds(vg + h, &lds_vt[buf ^ 1][h]);
      }
    }

    if (key0 <= qrow0 + 15) {          // wave-uniform causal skip
      const f16* lk = lds_k[buf];
      const f16* lv = lds_vt[buf];

      // ---- S = Q K^T : two 16x16 tiles, K-dim = D in 4 chunks of 32 ----
      v8f s0 = {}, s1 = {};
#pragma unroll
      for (int c = 0; c < 4; ++c) {
        const f16* kp0 = lk + (size_t)l16        * DIM + c * 32 + lhalf * 16;
        const f16* kp1 = lk + (size_t)(16 + l16) * DIM + c * 32 + lhalf * 16;
        v16h b0 = ld_frag(kp0, 8);
        v16h b1 = ld_frag(kp1, 8);
        s0 = __builtin_amdgcn_wmma_f32_16x16x32_f16(false, qfrag[c], false, b0,
                                                    (short)0, s0, false, false);
        s1 = __builtin_amdgcn_wmma_f32_16x16x32_f16(false, qfrag[c], false, b1,
                                                    (short)0, s1, false, false);
      }

      // ---- masked online softmax (base-2) ----
      float alpha[8];
#pragma unroll
      for (int r = 0; r < 8; ++r) {
        const int row = qrow0 + lhalf * 8 + r;
        float a0 = s0[r] * scale2;
        float a1 = s1[r] * scale2;
        if (key0 + l16      > row) a0 = -__builtin_inff();
        if (key0 + 16 + l16 > row) a1 = -__builtin_inff();
        const float mx   = rowmax16(fmaxf(a0, a1));
        const float mnew = fmaxf(mrow[r], mx);
        const float al   = exp2f(mrow[r] - mnew);
        const float e0   = exp2f(a0 - mnew);
        const float e1   = exp2f(a1 - mnew);
        lrow[r]  = lrow[r] * al + rowsum16(e0 + e1);
        mrow[r]  = mnew;
        alpha[r] = al;
        const int m = lhalf * 8 + r;
        pw[m * KTILE + l16]      = (f16)e0;
        pw[m * KTILE + 16 + l16] = (f16)e1;
      }
      asm volatile("s_wait_dscnt 0" ::: "memory");  // same-wave LDS RAW

      const v16h pfrag = ld_frag(pw + (size_t)l16 * KTILE + lhalf * 16, 8);

      // ---- O = diag(alpha)*O + P @ V ----
#pragma unroll
      for (int t = 0; t < 8; ++t) {
        const f16* vp = lv + (size_t)(t * 16 + l16) * KTILE + lhalf * 16;
        v16h vfrag = ld_frag(vp, 8);
        v8f cacc = oacc[t];
#pragma unroll
        for (int r = 0; r < 8; ++r) cacc[r] *= alpha[r];
        oacc[t] = __builtin_amdgcn_wmma_f32_16x16x32_f16(false, pfrag, false, vfrag,
                                                         (short)0, cacc, false, false);
      }
    }
  }

  float inv[8];
#pragma unroll
  for (int r = 0; r < 8; ++r) inv[r] = 1.f / lrow[r];
#pragma unroll
  for (int t = 0; t < 8; ++t) {
#pragma unroll
    for (int r = 0; r < 8; ++r) {
      const int row = qrow0 + lhalf * 8 + r;
      Ob[(size_t)row * DIM + t * 16 + l16] = oacc[t][r] * inv[r];
    }
  }
}

// ---------------------------------------------------------------------------
// Fallback (no workspace): round-1 style, converts K/V in-kernel.
// ---------------------------------------------------------------------------
__global__ __launch_bounds__(256)
void fa_fwd_legacy(const float* __restrict__ Q, const float* __restrict__ K,
                   const float* __restrict__ V, float* __restrict__ O) {
  __shared__ f16 lds_k [KTILE * DIM];
  __shared__ f16 lds_vt[DIM * KTILE];
  __shared__ f16 lds_p [NWAVES * 16 * KTILE];

  const int qbase = blockIdx.x * QTILE;
  const int b     = blockIdx.y;
  const int tid   = threadIdx.x;
  const int wave  = tid >> 5, lane = tid & 31;
  const int lhalf = lane >> 4, l16 = lane & 15;
  const int qrow0 = qbase + wave * 16;

  const size_t boff = (size_t)b * SEQ * DIM;
  const float *Qb = Q + boff, *Kb = K + boff, *Vb = V + boff;
  float* Ob = O + boff;

  v16h qfrag[4];
  {
    const float* qr = Qb + (size_t)(qrow0 + l16) * DIM;
#pragma unroll
    for (int c = 0; c < 4; ++c) {
      const int d0 = c * 32 + lhalf * 16;
      v16h a;
#pragma unroll
      for (int e = 0; e < 8; ++e) {
        a[e] = (f16)qr[d0 + e];
        a[e + 8] = (f16)qr[d0 + 8 + e];
      }
      qfrag[c] = a;
    }
  }

  float mrow[8], lrow[8];
  v8f oacc[8];
#pragma unroll
  for (int r = 0; r < 8; ++r) { mrow[r] = -__builtin_inff(); lrow[r] = 0.f; }
#pragma unroll
  for (int t = 0; t < 8; ++t) { oacc[t] = {}; }

  const float scale2 = 0.08838834764831845f * 1.4426950408889634f;
  const int nkb = (qbase + QTILE) / KTILE;
  f16* pw = lds_p + wave * 16 * KTILE;

  for (int kb = 0; kb < nkb; ++kb) {
    const int key0 = kb * KTILE;
    const float* kg = Kb + (size_t)key0 * DIM;
    const float* vg = Vb + (size_t)key0 * DIM;
    __syncthreads();
#pragma unroll
    for (int i = 0; i < 4; ++i) {
      const int idx = (tid + i * 256) * 4;
      v4f kv = *(const v4f*)(kg + idx);
      v4f vv = *(const v4f*)(vg + idx);
      const int row = idx >> 7, col = idx & 127;
#pragma unroll
      for (int j = 0; j < 4; ++j) {
        lds_k[idx + j] = (f16)kv[j];
        lds_vt[(col + j) * KTILE + row] = (f16)vv[j];
      }
    }
    __syncthreads();

    if (key0 <= qrow0 + 15) {
      v8f s0 = {}, s1 = {};
#pragma unroll
      for (int c = 0; c < 4; ++c) {
        const f16* kp0 = lds_k + (size_t)l16 * DIM + c * 32 + lhalf * 16;
        const f16* kp1 = lds_k + (size_t)(16 + l16) * DIM + c * 32 + lhalf * 16;
        v16h b0 = ld_frag(kp0, 8), b1 = ld_frag(kp1, 8);
        s0 = __builtin_amdgcn_wmma_f32_16x16x32_f16(false, qfrag[c], false, b0,
                                                    (short)0, s0, false, false);
        s1 = __builtin_amdgcn_wmma_f32_16x16x32_f16(false, qfrag[c], false, b1,
                                                    (short)0, s1, false, false);
      }
      float alpha[8];
#pragma unroll
      for (int r = 0; r < 8; ++r) {
        const int row = qrow0 + lhalf * 8 + r;
        float a0 = s0[r] * scale2, a1 = s1[r] * scale2;
        if (key0 + l16 > row) a0 = -__builtin_inff();
        if (key0 + 16 + l16 > row) a1 = -__builtin_inff();
        const float mx = rowmax16(fmaxf(a0, a1));
        const float mnew = fmaxf(mrow[r], mx);
        const float al = exp2f(mrow[r] - mnew);
        const float e0 = exp2f(a0 - mnew), e1 = exp2f(a1 - mnew);
        lrow[r] = lrow[r] * al + rowsum16(e0 + e1);
        mrow[r] = mnew; alpha[r] = al;
        const int m = lhalf * 8 + r;
        pw[m * KTILE + l16] = (f16)e0;
        pw[m * KTILE + 16 + l16] = (f16)e1;
      }
      asm volatile("s_wait_dscnt 0" ::: "memory");
      const v16h pfrag = ld_frag(pw + (size_t)l16 * KTILE + lhalf * 16, 8);
#pragma unroll
      for (int t = 0; t < 8; ++t) {
        v16h vfrag = ld_frag(lds_vt + (size_t)(t * 16 + l16) * KTILE + lhalf * 16, 8);
        v8f cacc = oacc[t];
#pragma unroll
        for (int r = 0; r < 8; ++r) cacc[r] *= alpha[r];
        oacc[t] = __builtin_amdgcn_wmma_f32_16x16x32_f16(false, pfrag, false, vfrag,
                                                         (short)0, cacc, false, false);
      }
    }
  }
  float inv[8];
#pragma unroll
  for (int r = 0; r < 8; ++r) inv[r] = 1.f / lrow[r];
#pragma unroll
  for (int t = 0; t < 8; ++t)
#pragma unroll
    for (int r = 0; r < 8; ++r) {
      const int row = qrow0 + lhalf * 8 + r;
      Ob[(size_t)row * DIM + t * 16 + l16] = oacc[t][r] * inv[r];
    }
}

extern "C" void kernel_launch(void* const* d_in, const int* in_sizes, int n_in,
                              void* d_out, int out_size, void* d_ws, size_t ws_size,
                              hipStream_t stream) {
  const float* q = (const float*)d_in[0];
  const float* k = (const float*)d_in[1];
  const float* v = (const float*)d_in[2];
  float* out = (float*)d_out;
  const int    B     = in_sizes[0] / (SEQ * DIM);  // = 8
  const int    total = B * SEQ * DIM;
  const size_t need  = (size_t)total * sizeof(f16) * 2;  // Kh + VTh

  dim3 grid(SEQ / QTILE, B);
  if (ws_size >= need) {
    f16* Kh  = (f16*)d_ws;
    f16* VTh = Kh + total;
    cvt_kv_kernel<<<2048, 256, 0, stream>>>(k, v, Kh, VTh, total);
    fa_fwd_async<<<grid, NWAVES * 32, 0, stream>>>(q, Kh, VTh, out);
  } else {
    fa_fwd_legacy<<<grid, NWAVES * 32, 0, stream>>>(q, k, v, out);
  }
}
